// PGN_12481174962836
// MI455X (gfx1250) — matrix operand order repaired and checked by reading, not compiled, not run on hardware
//
#include <hip/hip_runtime.h>
#include <hip/hip_bf16.h>

// ---------------------------------------------------------------------------
// PGN message-passing layer for MI455X (gfx1250).
// Dominant work: edge_fts[4,512,512,128] @ we.T fused with masked max over i.
// 34.4 GFLOP GEMM vs 536 MB streamed -> run on v_wmma_f32_16x16x32_bf16 so the
// kernel is HBM-bound (~23 us at 23.3 TB/s), never materializing msgs.
// ---------------------------------------------------------------------------

typedef __attribute__((ext_vector_type(16))) __bf16 v16bf;
typedef __attribute__((ext_vector_type(8)))  float  v8f;
typedef __attribute__((ext_vector_type(4)))  float  f32x4;   // native vector for NT loads

#define NB   4
#define NN   512
#define NM   128
#define K2   256
#define ICH  8          // i-chunks
#define CHUNK (NN / ICH)

// Fast f32x2 -> packed bf16x2: bias mantissa (round-half-up) + one v_perm_b32.
__device__ __forceinline__ unsigned int pk_bf16(float lo, float hi) {
    unsigned ul = __float_as_uint(lo) + 0x8000u;
    unsigned uh = __float_as_uint(hi) + 0x8000u;
    return __builtin_amdgcn_perm(uh, ul, 0x07060302u);  // {uh[31:16], ul[31:16]}
}

// ---------- P1a: msg1 = z@w1.T+b1 ; msg2 = z@w2.T+b2 ; h1z = z@ow1.T+ob1 ----
__global__ __launch_bounds__(128) void pgn_rowgemm(
    const float* __restrict__ node, const float* __restrict__ hidden,
    const float* __restrict__ w1, const float* __restrict__ b1,
    const float* __restrict__ w2, const float* __restrict__ b2,
    const float* __restrict__ ow1, const float* __restrict__ ob1,
    float* __restrict__ msg1, float* __restrict__ msg2, float* __restrict__ h1z)
{
    __shared__ __align__(16) float z[K2];
    const int bj = blockIdx.x;           // b*NN + j
    const int t  = threadIdx.x;
    z[t]       = node[bj * NM + t];
    z[NM + t]  = hidden[bj * NM + t];
    __syncthreads();

    const int o = t;
    float s1 = b1[o], s2 = b2[o], s3 = ob1[o];
    const float4* w1p  = (const float4*)(w1  + (size_t)o * K2);
    const float4* w2p  = (const float4*)(w2  + (size_t)o * K2);
    const float4* ow1p = (const float4*)(ow1 + (size_t)o * K2);
    #pragma unroll 8
    for (int q = 0; q < K2 / 4; ++q) {
        float4 zv = *(const float4*)(z + 4 * q);
        float4 a  = w1p[q], bq = w2p[q], c = ow1p[q];
        s1 += zv.x * a.x  + zv.y * a.y  + zv.z * a.z  + zv.w * a.w;
        s2 += zv.x * bq.x + zv.y * bq.y + zv.z * bq.z + zv.w * bq.w;
        s3 += zv.x * c.x  + zv.y * c.y  + zv.z * c.z  + zv.w * c.w;
    }
    msg1[bj * NM + o] = s1;
    msg2[bj * NM + o] = s2;
    h1z [bj * NM + o] = s3;
}

// ---------- P1b: red = -1e6 ; webf = bf16(we) ; msgg = graph@wg.T+bg --------
__global__ __launch_bounds__(256) void pgn_prep(
    const float* __restrict__ we, const float* __restrict__ graph,
    const float* __restrict__ wg, const float* __restrict__ bg,
    float* __restrict__ red, unsigned short* __restrict__ webf,
    float* __restrict__ msgg)
{
    const int idx = blockIdx.x * 256 + threadIdx.x;   // 0 .. 262143
    red[idx] = -1000000.0f;
    if (idx < NM * NM) {
        // weights: keep proper round-to-nearest-even
        unsigned u = __float_as_uint(we[idx]);
        u = (u + 0x7FFFu + ((u >> 16) & 1u)) >> 16;
        webf[idx] = (unsigned short)u;
    }
    if (idx < NB * NM) {
        const int b = idx / NM, o = idx % NM;
        float s = bg[o];
        const float4* gp = (const float4*)(graph + (size_t)b * NM);
        const float4* wp = (const float4*)(wg + (size_t)o * NM);
        #pragma unroll 8
        for (int q = 0; q < NM / 4; ++q) {
            float4 g = gp[q], w = wp[q];
            s += g.x * w.x + g.y * w.y + g.z * w.z + g.w * w.w;
        }
        msgg[idx] = s;
    }
}

// ---------- P2: fused msg_e WMMA + broadcast adds + masked max over i -------
__global__ __launch_bounds__(128) void pgn_main(
    const float* __restrict__ edge, const float* __restrict__ adj,
    const float* __restrict__ msg1, const float* __restrict__ msg2,
    const float* __restrict__ msgg, const float* __restrict__ be,
    const unsigned short* __restrict__ webf, float* __restrict__ red)
{
    __shared__ __align__(16) uint4 ldsA[2][16 * 16];   // 16 rows x 128 bf16, dbl-buffered

    const int bid = blockIdx.x;            // [0, 4*32*8)
    const int ic  = bid & (ICH - 1);
    const int jt  = (bid >> 3) & 31;
    const int b   = bid >> 8;
    const int j0  = jt * 16;
    const int i0  = ic * CHUNK;

    const int t    = threadIdx.x;
    const int lane = t & 31;
    const int wv   = t >> 5;               // wave 0..3 -> owns cols 32*wv .. +31
    const int lrow = lane & 15;
    const int sel  = lane >> 4;            // lane half

    // --- B fragments (we.T in bf16), resident for the whole kernel ---------
    v16bf Bf[2][4];
    int col[2];
    #pragma unroll
    for (int tt = 0; tt < 2; ++tt) {
        col[tt] = (wv * 2 + tt) * 16 + lrow;        // N index (output feature)
        #pragma unroll
        for (int c = 0; c < 4; ++c) {               // K chunk of 32
            union { uint4 u[2]; v16bf v; } tmp;
            const uint4* p = (const uint4*)(webf + (size_t)col[tt] * NM + c * 32 + sel * 16);
            tmp.u[0] = p[0]; tmp.u[1] = p[1];
            Bf[tt][c] = tmp.v;
        }
    }

    // --- C-base: msg1[b,j,col] + msg_g[b,col] + be[col]; acc = -1e6 --------
    float base[2][8], accv[2][8];
    #pragma unroll
    for (int tt = 0; tt < 2; ++tt) {
        const float g = msgg[b * NM + col[tt]] + be[col[tt]];
        #pragma unroll
        for (int v = 0; v < 8; ++v) {
            const int j = j0 + v + 8 * sel;
            base[tt][v] = msg1[((size_t)b * NN + j) * NM + col[tt]] + g;
            accv[tt][v] = -1000000.0f;
        }
    }

    const int srow = t >> 3, skg = t & 7;  // staging: row 0..15, k-group 0..7

    auto stage = [&](int i, int buf) {
        const float* sp = edge + (((size_t)b * NN + i) * NN + (j0 + srow)) * NM + skg * 16;
        f32x4 f0 = __builtin_nontemporal_load((const f32x4*)sp + 0);
        f32x4 f1 = __builtin_nontemporal_load((const f32x4*)sp + 1);
        f32x4 f2 = __builtin_nontemporal_load((const f32x4*)sp + 2);
        f32x4 f3 = __builtin_nontemporal_load((const f32x4*)sp + 3);
        uint4 p0 = make_uint4(pk_bf16(f0.x, f0.y), pk_bf16(f0.z, f0.w),
                              pk_bf16(f1.x, f1.y), pk_bf16(f1.z, f1.w));
        uint4 p1 = make_uint4(pk_bf16(f2.x, f2.y), pk_bf16(f2.z, f2.w),
                              pk_bf16(f3.x, f3.y), pk_bf16(f3.z, f3.w));
        ldsA[buf][srow * 16 + skg * 2 + 0] = p0;
        ldsA[buf][srow * 16 + skg * 2 + 1] = p1;
    };

    stage(i0, 0);

    for (int ii = 0; ii < CHUNK; ++ii) {
        __syncthreads();
        if (ii + 1 < CHUNK) stage(i0 + ii + 1, (ii + 1) & 1);
        const uint4* abuf = &ldsA[ii & 1][lrow * 16];
        const int i = i0 + ii;

        // adjacency: additive mask bias (0 if neighbor, -4e6 if not; the -4e6
        // branch can never beat the -1e6 accumulator init, matching where()).
        const float* ap = adj + ((size_t)b * NN + i) * NN + j0 + sel * 8;
        f32x4 a0 = __builtin_nontemporal_load((const f32x4*)ap + 0);
        f32x4 a1 = __builtin_nontemporal_load((const f32x4*)ap + 1);
        float adjv[8] = {a0.x, a0.y, a0.z, a0.w, a1.x, a1.y, a1.z, a1.w};
        float bias[8];
        #pragma unroll
        for (int v = 0; v < 8; ++v)
            bias[v] = __builtin_fmaf(adjv[v], 4.0e6f, -4.0e6f);

        v8f C[2];
        #pragma unroll
        for (int tt = 0; tt < 2; ++tt) {
            const float m2v = msg2[((size_t)b * NN + i) * NM + col[tt]];
            #pragma unroll
            for (int v = 0; v < 8; ++v) C[tt][v] = base[tt][v] + m2v;
        }

        #pragma unroll
        for (int c = 0; c < 4; ++c) {
            union { uint4 u[2]; v16bf v; } af;
            af.u[0] = abuf[4 * c + 2 * sel];
            af.u[1] = abuf[4 * c + 2 * sel + 1];
            C[0] = __builtin_amdgcn_wmma_f32_16x16x32_bf16(
                       false, af.v, false, Bf[0][c], (short)0, C[0], false, false);
            C[1] = __builtin_amdgcn_wmma_f32_16x16x32_bf16(
                       false, af.v, false, Bf[1][c], (short)0, C[1], false, false);
        }

        #pragma unroll
        for (int tt = 0; tt < 2; ++tt)
            #pragma unroll
            for (int v = 0; v < 8; ++v) {
                const float m = C[tt][v] + bias[v];
                accv[tt][v] = (m > accv[tt][v]) ? m : accv[tt][v];
            }
    }

    // --- combine chunks: order-preserving float max via integer atomics ----
    #pragma unroll
    for (int tt = 0; tt < 2; ++tt) {
        #pragma unroll
        for (int v = 0; v < 8; ++v) {
            const float val = accv[tt][v];
            float* addr = red + ((size_t)b * NN + j0 + v + 8 * sel) * NM + col[tt];
            if (val >= 0.0f) atomicMax((int*)addr, __float_as_int(val));
            else             atomicMin((unsigned int*)addr, __float_as_uint(val));
        }
    }
}

// ---------- P3: out = relu(h1z + red@ow2.T + ob2) ---------------------------
__global__ __launch_bounds__(128) void pgn_out(
    const float* __restrict__ red, const float* __restrict__ h1z,
    const float* __restrict__ ow2, const float* __restrict__ ob2,
    float* __restrict__ out)
{
    __shared__ __align__(16) float rrow[NM];
    const int bj = blockIdx.x;
    const int o  = threadIdx.x;
    rrow[o] = red[bj * NM + o];
    __syncthreads();

    float s = h1z[bj * NM + o] + ob2[o];
    const float4* wp = (const float4*)(ow2 + (size_t)o * NM);
    #pragma unroll 8
    for (int q = 0; q < NM / 4; ++q) {
        float4 r = *(const float4*)(rrow + 4 * q);
        float4 w = wp[q];
        s += r.x * w.x + r.y * w.y + r.z * w.z + r.w * w.w;
    }
    out[bj * NM + o] = fmaxf(s, 0.0f);
}

// ---------------------------------------------------------------------------
extern "C" void kernel_launch(void* const* d_in, const int* in_sizes, int n_in,
                              void* d_out, int out_size, void* d_ws, size_t ws_size,
                              hipStream_t stream) {
    const float* node   = (const float*)d_in[0];
    const float* edge   = (const float*)d_in[1];
    const float* graph  = (const float*)d_in[2];
    const float* adj    = (const float*)d_in[3];
    const float* hidden = (const float*)d_in[4];
    const float* w1  = (const float*)d_in[5];
    const float* b1  = (const float*)d_in[6];
    const float* w2  = (const float*)d_in[7];
    const float* b2  = (const float*)d_in[8];
    const float* we  = (const float*)d_in[9];
    const float* be  = (const float*)d_in[10];
    const float* wg  = (const float*)d_in[11];
    const float* bg  = (const float*)d_in[12];
    const float* ow1 = (const float*)d_in[13];
    const float* ob1 = (const float*)d_in[14];
    const float* ow2 = (const float*)d_in[15];
    const float* ob2 = (const float*)d_in[16];
    float* out = (float*)d_out;

    const size_t BNM = (size_t)NB * NN * NM;         // 262144
    float* msg1 = (float*)d_ws;
    float* msg2 = msg1 + BNM;
    float* h1z  = msg2 + BNM;
    float* red  = h1z  + BNM;
    float* msgg = red  + BNM;                        // NB*NM = 512
    unsigned short* webf = (unsigned short*)(msgg + NB * NM);  // 16384 bf16

    pgn_rowgemm<<<NB * NN, 128, 0, stream>>>(node, hidden, w1, b1, w2, b2,
                                             ow1, ob1, msg1, msg2, h1z);
    pgn_prep<<<(int)(BNM / 256), 256, 0, stream>>>(we, graph, wg, bg,
                                                   red, webf, msgg);
    pgn_main<<<NB * 32 * ICH, 128, 0, stream>>>(edge, adj, msg1, msg2, msgg,
                                                be, webf, red);
    pgn_out<<<NB * NN, 128, 0, stream>>>(red, h1z, ow2, ob2, out);
}
